// FastGroupedAttention_13486197310068
// MI455X (gfx1250) — matrix operand order repaired and checked by reading, not compiled, not run on hardware
//
#include <hip/hip_runtime.h>
#include <hip/hip_bf16.h>

// FastGroupedAttention for MI455X (gfx1250, wave32, WMMA + TDM).
// Pipeline: cvt(x,in_w,out_w -> bf16) ; GEMM1(bf16 wmma, TDM-staged) -> packed QKV ;
// flash-attention (bf16 wmma QK^T + PV, fp32 online softmax) -> y ;
// GEMM2(bf16 wmma, TDM-staged) -> fp32 out.
// Workspace requirement: ~76 MB (xb 16MB + wb 12MB + owb 8MB + packed 24MB + y 16MB).

typedef __bf16 bf16_t;
typedef __attribute__((ext_vector_type(16))) __bf16 v16bf;
typedef __attribute__((ext_vector_type(8)))  float  v8f;

#if __has_builtin(__builtin_amdgcn_tensor_load_to_lds) && \
    __has_builtin(__builtin_amdgcn_s_wait_tensorcnt)
#define USE_TDM 1
#else
#define USE_TDM 0
#endif

#if USE_TDM
typedef __attribute__((ext_vector_type(4))) unsigned int v4u;
typedef __attribute__((ext_vector_type(8))) int v8i_;
typedef __attribute__((ext_vector_type(4))) int v4i_;

// Issue a TDM 2D tile load: tile_d1 rows x tile_d0 bf16 elements, global row
// stride `row_stride` elements, destination LDS offset `lds_off` with TDM
// padding (pad_interval: 2^(v+1) DWORDs between pads, pad_amount: v+1 DWORDs).
// D# layout per CDNA5 ISA 8.3/8.4 (group0 128b, group1 256b; remaining groups
// zero for 2D tensors). tensor_dim == tile_dim so no OOB clipping occurs.
__device__ __forceinline__ void tdm_load_tile_bf16(unsigned lds_off,
                                                   const bf16_t* gaddr,
                                                   int tile_d0, int tile_d1,
                                                   long row_stride,
                                                   int pad_interval,
                                                   int pad_amount) {
  const unsigned long long ga = (unsigned long long)gaddr;
  v4u g0;
  g0[0] = 1u;                                   // count=1, user descriptor
  g0[1] = lds_off;                              // lds_addr (bytes)
  g0[2] = (unsigned)ga;                         // global_addr[31:0]
  g0[3] = (unsigned)((ga >> 32) & 0x1ffffffull) // global_addr[56:32]
          | (2u << 30);                         // type=2 ("image")
  v8i_ g1;
  g1[0] = (1 << 16)                             // data_size=1 -> 2 bytes
          | (1 << 20)                           // pad_enable
          | (pad_interval << 22) | (pad_amount << 25);
  g1[1] = (tile_d0 & 0xffff) << 16;             // tensor_dim0[15:0]
  g1[2] = ((tile_d0 >> 16) & 0xffff)            // tensor_dim0[31:16]
          | ((tile_d1 & 0xffff) << 16);         // tensor_dim1[15:0]
  g1[3] = ((tile_d1 >> 16) & 0xffff)            // tensor_dim1[31:16]
          | ((tile_d0 & 0xffff) << 16);         // tile_dim0
  g1[4] = (tile_d1 & 0xffff);                   // tile_dim1 (tile_dim2=0)
  g1[5] = (int)(row_stride & 0xffffffffl);      // tensor_dim0_stride[31:0]
  g1[6] = (int)((row_stride >> 32) & 0xffffl);  // tensor_dim0_stride[47:32]
  g1[7] = 0;                                    // tensor_dim1_stride=0 (2D)
  const v4i_ z4 = {0, 0, 0, 0};
  const v8i_ z8 = {0, 0, 0, 0, 0, 0, 0, 0};
  // clang-23 / therock-10.0 6-arg form: (g0, g1, g2, g3, gx, cpol)
  __builtin_amdgcn_tensor_load_to_lds(g0, g1, z4, z4, z8, 0);
}
#endif

// ---------------------------------------------------------------------------
// WMMA fragment loaders (layouts per CDNA5 ISA 7.12.2, wave32)
// A 16x32 bf16: lanes 0-15 row=lane, elems 0..7 -> K 0..7, 8..15 -> K 16..23
//               lanes 16-31 row=lane-16, elems 0..7 -> K 8..15, 8..15 -> K 24..31
__device__ __forceinline__ v16bf wmma_load_a(const bf16_t* p0, int ld, int kbase) {
  const int lane = threadIdx.x & 31;
  const int hl = lane >> 4;
  const int m  = lane & 15;
  const bf16_t* p = p0 + m * ld + kbase;
  v16bf a;
#pragma unroll
  for (int j = 0; j < 8; ++j) a[j] = p[8 * hl + j];
#pragma unroll
  for (int j = 0; j < 8; ++j) a[8 + j] = p[16 + 8 * hl + j];
  return a;
}

// B 32x16 bf16 loaded from a [N][K] (row-major B-transpose) tile:
// lane: N = lane%16, elems 0..7 -> K = kbase + 8*(lane/16) + j, 8..15 -> K = kbase + 16 + 8*(lane/16) + (j-8)
__device__ __forceinline__ v16bf wmma_load_bt(const bf16_t* p0, int ld, int kbase) {
  const int lane = threadIdx.x & 31;
  const int hl = lane >> 4;
  const int n  = lane & 15;
  const bf16_t* p = p0 + n * ld + kbase;
  v16bf b;
#pragma unroll
  for (int j = 0; j < 8; ++j) b[j] = p[8 * hl + j];
#pragma unroll
  for (int j = 0; j < 8; ++j) b[8 + j] = p[16 + 8 * hl + j];
  return b;
}

__device__ __forceinline__ v8f wmma_bf16(v16bf a, v16bf b, v8f c) {
  // (neg_a, A, neg_b, B, c_mod, C, reuse_a, reuse_b)
  return __builtin_amdgcn_wmma_f32_16x16x32_bf16(false, a, false, b, (short)0, c,
                                                 false, false);
}

// ---------------------------------------------------------------------------
__global__ __launch_bounds__(256)
void cvt_f32_to_bf16(const float* __restrict__ s, bf16_t* __restrict__ d, int n4) {
  const int i = blockIdx.x * 256 + threadIdx.x;
  if (i < n4) {
    float4 f = ((const float4*)s)[i];
    d[4 * i + 0] = (bf16_t)f.x;
    d[4 * i + 1] = (bf16_t)f.y;
    d[4 * i + 2] = (bf16_t)f.z;
    d[4 * i + 3] = (bf16_t)f.w;
  }
}

// ---------------------------------------------------------------------------
// C[M x N] = A[M x K] * Bt[N x K]^T + bias[N]
// Block tile 128x128, K step 32, 256 threads = 8 waves (2x4), wave tile 64x32.
// Tiles staged in LDS by the Tensor Data Mover (wave 0 issues descriptors),
// double-buffered so DMA of stage i+1 overlaps WMMA of stage i.
template <bool F32OUT>
__global__ __launch_bounds__(256)
void gemm_bt_bf16(const bf16_t* __restrict__ A, const bf16_t* __restrict__ Bt,
                  const float* __restrict__ bias, void* __restrict__ Cout,
                  int K, int ldc) {
  constexpr int LDT = 40;  // 32 + 8 bf16 pad; TDM pad 4 DWORDs / 16 DWORDs row
  __shared__ bf16_t sA[2][128 * LDT];
  __shared__ bf16_t sB[2][128 * LDT];

  const int tid = threadIdx.x;
  const int wave = tid >> 5, lane = tid & 31;
  const int wm = wave >> 2, wn = wave & 3;
  const int m0 = blockIdx.y * 128, n0 = blockIdx.x * 128;

  v8f acc[4][2];
#pragma unroll
  for (int i = 0; i < 4; ++i)
#pragma unroll
    for (int j = 0; j < 2; ++j) acc[i][j] = {};

  const bf16_t* gA0 = A + (size_t)m0 * K;
  const bf16_t* gB0 = Bt + (size_t)n0 * K;
  const int nk = K / 32;

#if USE_TDM
  const unsigned ldsA[2] = {(unsigned)(size_t)(void*)&sA[0][0],
                            (unsigned)(size_t)(void*)&sA[1][0]};
  const unsigned ldsB[2] = {(unsigned)(size_t)(void*)&sB[0][0],
                            (unsigned)(size_t)(void*)&sB[1][0]};
  // Prime stage 0. TDM ignores EXEC but issues once per executing wave, so
  // only wave 0 (uniform branch) issues descriptors.
  if (wave == 0) {
    tdm_load_tile_bf16(ldsA[0], gA0, 32, 128, K, 3, 3);
    tdm_load_tile_bf16(ldsB[0], gB0, 32, 128, K, 3, 3);
  }
#else
  const int lr = tid >> 1, lc = (tid & 1) * 16;
  const bf16_t* ga = A + (size_t)(m0 + lr) * K + lc;
  const bf16_t* gb = Bt + (size_t)(n0 + lr) * K + lc;
#endif

  for (int i = 0; i < nk; ++i) {
    const int cur = i & 1;
#if USE_TDM
    if (wave == 0) {
      if (i + 1 < nk) {
        // Prefetch next stage into the other buffer, then wait for the two
        // oldest (current stage) DMAs: TDM ops of one wave finish in order.
        tdm_load_tile_bf16(ldsA[cur ^ 1], gA0 + (size_t)(i + 1) * 32, 32, 128, K, 3, 3);
        tdm_load_tile_bf16(ldsB[cur ^ 1], gB0 + (size_t)(i + 1) * 32, 32, 128, K, 3, 3);
        __builtin_amdgcn_s_wait_tensorcnt(2);
      } else {
        __builtin_amdgcn_s_wait_tensorcnt(0);
      }
    }
    __syncthreads();  // current-stage tiles visible to all waves
#else
    __syncthreads();
    *(uint4*)&sA[cur][lr * LDT + lc]     = *(const uint4*)(ga + i * 32);
    *(uint4*)&sA[cur][lr * LDT + lc + 8] = *(const uint4*)(ga + i * 32 + 8);
    *(uint4*)&sB[cur][lr * LDT + lc]     = *(const uint4*)(gb + i * 32);
    *(uint4*)&sB[cur][lr * LDT + lc + 8] = *(const uint4*)(gb + i * 32 + 8);
    __syncthreads();
#endif

    v16bf bfr[2];
#pragma unroll
    for (int j = 0; j < 2; ++j)
      bfr[j] = wmma_load_bt(&sB[cur][(wn * 32 + j * 16) * LDT], LDT, 0);
#pragma unroll
    for (int ii = 0; ii < 4; ++ii) {
      v16bf af = wmma_load_a(&sA[cur][(wm * 64 + ii * 16) * LDT], LDT, 0);
#pragma unroll
      for (int j = 0; j < 2; ++j) acc[ii][j] = wmma_bf16(af, bfr[j], acc[ii][j]);
    }
#if USE_TDM
    __syncthreads();  // all reads of `cur` done before it is re-filled
#endif
  }

  const int hl = lane >> 4, ln = lane & 15;
#pragma unroll
  for (int i = 0; i < 4; ++i)
#pragma unroll
    for (int j = 0; j < 2; ++j) {
      const int col = n0 + wn * 32 + j * 16 + ln;
      const float bv = bias[col];
#pragma unroll
      for (int r = 0; r < 8; ++r) {
        const int row = m0 + wm * 64 + i * 16 + r + 8 * hl;
        const float v = acc[i][j][r] + bv;
        if (F32OUT) ((float*)Cout)[(size_t)row * ldc + col] = v;
        else        ((bf16_t*)Cout)[(size_t)row * ldc + col] = (bf16_t)v;
      }
    }
}

// ---------------------------------------------------------------------------
// Flash attention. Grid (t/64, 16 heads, b). Block = 128 threads = 4 waves.
// Wave w owns query rows [qb*64 + 16w, +16). Streams 64-key blocks.
__global__ __launch_bounds__(128)
void attn_fwd(const bf16_t* __restrict__ packed, bf16_t* __restrict__ y) {
  constexpr int T = 2048, LDPK = 3072, HD = 128, QB = 64, KB = 64;
  constexpr int LDK = HD + 8;  // sK  row stride (bf16)
  constexpr int LDV = KB + 8;  // sVt row stride (bf16)
  constexpr int LDS_ = KB + 4; // sS  row stride (f32)
  constexpr int LDPP = KB + 8; // sP  row stride (bf16)
  __shared__ bf16_t sK[KB * LDK];    // [key][d]   == Bt tile for QK^T
  __shared__ bf16_t sVt[HD * LDV];   // [d][key]   == Bt tile for P*V
  __shared__ float  sS[4 * 16 * LDS_];
  __shared__ bf16_t sP[4 * 16 * LDPP];
  __shared__ float  sStat[4 * 16];
  static_assert(sizeof(sK) + sizeof(sVt) + sizeof(sS) + sizeof(sP) + sizeof(sStat)
                    <= 64 * 1024, "LDS budget");

  const int tid = threadIdx.x, wave = tid >> 5, lane = tid & 31;
  const int hl = lane >> 4, ln = lane & 15;
  const int qb = blockIdx.x, h = blockIdx.y, bb = blockIdx.z;
  const int kv = h >> 2;  // head h -> kv group h/KV_REPEAT
  const size_t rowb = (size_t)bb * T;

  // Q fragments (16 rows x 128 head-dim) kept in registers for the whole block.
  v16bf qf[4];
  {
    const bf16_t* q0p = packed + (rowb + qb * QB + wave * 16) * LDPK + h * HD;
#pragma unroll
    for (int kk = 0; kk < 4; ++kk) qf[kk] = wmma_load_a(q0p, LDPK, kk * 32);
  }

  v8f oacc[8];
#pragma unroll
  for (int d = 0; d < 8; ++d) oacc[d] = {};
  float m_i = -1e30f, l_i = 0.f;
  const float scale = 0.088388347648318447f;  // 1/sqrt(128)

  const int lr = tid >> 1;        // 0..63 key row for staging
  const int lc = (tid & 1) * 64;  // d/key segment

  float*  sSw = sS + wave * 16 * LDS_;
  bf16_t* sPw = sP + wave * 16 * LDPP;

  for (int kb = 0; kb < T / KB; ++kb) {
    __syncthreads();
    const bf16_t* kg = packed + (rowb + kb * KB + lr) * LDPK + 2048 + kv * HD + lc;
    const bf16_t* vg = packed + (rowb + kb * KB + lr) * LDPK + 2560 + kv * HD + lc;
#pragma unroll
    for (int q8 = 0; q8 < 8; ++q8)
      *(uint4*)&sK[lr * LDK + lc + q8 * 8] = *(const uint4*)(kg + q8 * 8);
#pragma unroll
    for (int c = 0; c < 64; ++c)  // transpose V into [d][key]
      sVt[(lc + c) * LDV + lr] = vg[c];
    __syncthreads();

    // S = Q * K^T  (16 x 64 per wave)
    v8f sf[4];
#pragma unroll
    for (int ns = 0; ns < 4; ++ns) {
      v8f a = {};
#pragma unroll
      for (int kk = 0; kk < 4; ++kk) {
        v16bf bfK = wmma_load_bt(&sK[(ns * 16) * LDK], LDK, kk * 32);
        a = wmma_bf16(qf[kk], bfK, a);
      }
      sf[ns] = a;
    }
#pragma unroll
    for (int ns = 0; ns < 4; ++ns)
#pragma unroll
      for (int r = 0; r < 8; ++r)
        sSw[(r + 8 * hl) * LDS_ + ns * 16 + ln] = sf[ns][r] * scale;
    __syncthreads();

    // Online softmax: 2 lanes per row, 32 columns each.
    const int row = lane >> 1, cs = (lane & 1) * 32;
    const float* srow = sSw + row * LDS_ + cs;
    float mx = -1e30f;
#pragma unroll
    for (int c = 0; c < 32; ++c) mx = fmaxf(mx, srow[c]);
    mx = fmaxf(mx, __shfl_xor(mx, 1, 32));
    const float m_new = fmaxf(m_i, mx);
    const float alpha = __expf(m_i - m_new);
    bf16_t* prow = sPw + row * LDPP + cs;
    float sum = 0.f;
#pragma unroll
    for (int c = 0; c < 32; ++c) {
      const float p = __expf(srow[c] - m_new);
      prow[c] = (bf16_t)p;
      sum += p;
    }
    sum += __shfl_xor(sum, 1, 32);
    m_i = m_new;
    l_i = l_i * alpha + sum;
    if (!(lane & 1)) sStat[wave * 16 + row] = alpha;
    __syncthreads();

    // Rescale O and accumulate P*V (16 x 128 per wave).
    float arow[8];
#pragma unroll
    for (int r = 0; r < 8; ++r) arow[r] = sStat[wave * 16 + r + 8 * hl];
    v16bf pf[2];
#pragma unroll
    for (int kk = 0; kk < 2; ++kk) pf[kk] = wmma_load_a(sPw, LDPP, kk * 32);
#pragma unroll
    for (int ds = 0; ds < 8; ++ds) {
#pragma unroll
      for (int r = 0; r < 8; ++r) oacc[ds][r] *= arow[r];
#pragma unroll
      for (int kk = 0; kk < 2; ++kk) {
        v16bf bfV = wmma_load_bt(&sVt[(ds * 16) * LDV], LDV, kk * 32);
        oacc[ds] = wmma_bf16(pf[kk], bfV, oacc[ds]);
      }
    }
  }

  // Epilogue: O /= l, write y[b, t, h*128 + d].
  if (!(lane & 1)) sStat[wave * 16 + (lane >> 1)] = l_i;
  __syncthreads();
  float linv[8];
#pragma unroll
  for (int r = 0; r < 8; ++r) linv[r] = 1.f / sStat[wave * 16 + r + 8 * hl];
  bf16_t* yb = y + (rowb + qb * QB + wave * 16) * 2048 + h * HD;
#pragma unroll
  for (int ds = 0; ds < 8; ++ds)
#pragma unroll
    for (int r = 0; r < 8; ++r)
      yb[(size_t)(r + 8 * hl) * 2048 + ds * 16 + ln] = (bf16_t)(oacc[ds][r] * linv[r]);
}

// ---------------------------------------------------------------------------
extern "C" void kernel_launch(void* const* d_in, const int* in_sizes, int n_in,
                              void* d_out, int out_size, void* d_ws, size_t ws_size,
                              hipStream_t stream) {
  (void)in_sizes; (void)n_in; (void)out_size; (void)ws_size;
  const float* x     = (const float*)d_in[0];   // [2,2048,2048]
  const float* in_w  = (const float*)d_in[1];   // [3072,2048]
  const float* in_b  = (const float*)d_in[2];   // [3072]
  const float* out_w = (const float*)d_in[3];   // [2048,2048]
  const float* out_b = (const float*)d_in[4];   // [2048]
  float* out = (float*)d_out;                   // [2,2048,2048] f32

  bf16_t* ws = (bf16_t*)d_ws;
  bf16_t* xb     = ws; ws += (size_t)4096 * 2048;
  bf16_t* wb     = ws; ws += (size_t)3072 * 2048;
  bf16_t* owb    = ws; ws += (size_t)2048 * 2048;
  bf16_t* packed = ws; ws += (size_t)4096 * 3072;
  bf16_t* yb     = ws;

  cvt_f32_to_bf16<<<(4096 * 2048) / 1024, 256, 0, stream>>>(x, xb, (4096 * 2048) / 4);
  cvt_f32_to_bf16<<<(3072 * 2048) / 1024, 256, 0, stream>>>(in_w, wb, (3072 * 2048) / 4);
  cvt_f32_to_bf16<<<(2048 * 2048) / 1024, 256, 0, stream>>>(out_w, owb, (2048 * 2048) / 4);

  // packed[4096,3072] = xb * in_w^T + in_b
  gemm_bt_bf16<false><<<dim3(3072 / 128, 4096 / 128), 256, 0, stream>>>(
      xb, wb, in_b, (void*)packed, 2048, 3072);

  // y[4096,2048] = attention(packed)
  attn_fwd<<<dim3(2048 / 64, 16, 2), 128, 0, stream>>>(packed, yb);

  // out[4096,2048] = y * out_w^T + out_b   (f32)
  gemm_bt_bf16<true><<<dim3(2048 / 128, 4096 / 128), 256, 0, stream>>>(
      yb, owb, out_b, (void*)out, 2048, 2048);
}
